// MultiHeadAttention_62259845923177
// MI455X (gfx1250) — compile-verified, matrix operand order checked by reading, not run on hardware
//
#include <hip/hip_runtime.h>
#include <hip/hip_bf16.h>

// MI455X (gfx1250) MHA forward. Heavy math via v_wmma_f32_16x16x32_bf16.
// Attention K-tiles staged by the Tensor Data Mover (tensor_load_to_lds),
// V-tiles by async global->LDS copies (ASYNCcnt), per CDNA5 ISA.
//
// Workspace layout (bytes into d_ws), total 88 MB:
//   [0,16M)   Xb   : x as bf16           [8192 x 1024]
//   [16M,18M) Wqb, [18M,20M) Wkb, [20M,22M) Wvb, [22M,24M) Wob
//   [24M,40M) Qt, [40M,56M) Kt, [56M,72M) Vt   : bf16 [B,H,S,D]
//   [72M,88M) Ob   : attention output bf16 [8192 x 1024]

#define DI __device__ __forceinline__

typedef __bf16 bf16;
typedef __attribute__((ext_vector_type(16))) __bf16 v16bf;
typedef __attribute__((ext_vector_type(8)))  __bf16 v8bf;
typedef __attribute__((ext_vector_type(8)))  float  v8f;
typedef __attribute__((ext_vector_type(4)))  unsigned int u32x4;
typedef __attribute__((ext_vector_type(8)))  unsigned int u32x8;

constexpr int E = 1024, NH = 16, HD = 64, SEQ = 2048, BATCH = 4;
constexpr int MTOT = BATCH * SEQ;  // 8192

DI v8f wmma_bf16(v16bf a, v16bf b, v8f c) {
  return __builtin_amdgcn_wmma_f32_16x16x32_bf16(
      /*neg_a=*/false, a, /*neg_b=*/false, b,
      /*c_mod=*/(short)0, c, /*reuse_a=*/false, /*reuse_b=*/false);
}

// Load one 16x32 bf16 A-fragment (or B-fragment as rows of B^T) from a
// row-major matrix. ISA 16-bit A layout: lanes 0-15 -> M=lane, elems 0..7 are
// K=k0..k0+7, elems 8..15 are K=k0+16..k0+23; lanes 16-31 are K offset +8.
DI v16bf load_frag(const bf16* __restrict__ base, int ld, int rowbase, int k0) {
  const int lane = threadIdx.x & 31;
  const int half = lane >> 4;
  const int row  = rowbase + (lane & 15);
  const bf16* p = base + row * ld + k0 + half * 8;
  v8bf lo = *(const v8bf*)(p);
  v8bf hi = *(const v8bf*)(p + 16);
  v16bf r;
#pragma unroll
  for (int i = 0; i < 8; ++i) { r[i] = lo[i]; r[i + 8] = hi[i]; }
  return r;
}

// --- CDNA5 async / TDM helpers ------------------------------------------

// GVS-mode async copy: LDS[lds_off] <- MEM[sbase + voff], 16 bytes per lane.
DI void async_load_b128(unsigned int lds_off, unsigned int voff,
                        unsigned long long sbase) {
  asm volatile("global_load_async_to_lds_b128 %0, %1, %2"
               :: "v"(lds_off), "v"(voff), "s"(sbase) : "memory");
}

DI void wait_asynccnt0() { asm volatile("s_wait_asynccnt 0" ::: "memory"); }

// TDM 2D tile load: LDS[lds_off] <- 2-byte elements, tile (tile0 x tile1),
// row stride stride0 (elements). D# layout per CDNA5 ISA.
DI void tdm_load_2d(unsigned int lds_off, const void* gptr,
                    unsigned int tdim0, unsigned int tdim1,
                    unsigned int tile0, unsigned int tile1,
                    unsigned int stride0) {
  unsigned long long ga = (unsigned long long)gptr;
  u32x4 g0;
  g0[0] = 1u;                                   // count=1, user-mode D#
  g0[1] = lds_off;                              // lds_addr (bytes)
  g0[2] = (unsigned int)ga;                     // global_addr[31:0]
  g0[3] = ((unsigned int)(ga >> 32) & 0x01FFFFFFu) | (2u << 30);  // [56:32],type=2
  u32x8 g1;
  g1[0] = 1u << 16;                             // wg_mask=0, data_size=1 (2B)
  g1[1] = (tdim0 & 0xFFFFu) << 16;              // tensor_dim0[15:0]
  g1[2] = (tdim0 >> 16) | ((tdim1 & 0xFFFFu) << 16);  // td0[31:16]|td1[15:0]
  g1[3] = (tdim1 >> 16) | (tile0 << 16);        // td1[31:16] | tile_dim0
  g1[4] = tile1;                                // tile_dim1 | tile_dim2=0
  g1[5] = stride0;                              // tensor_dim0_stride[31:0]
  g1[6] = 0;                                    // stride0[47:32]|stride1[15:0]
  g1[7] = 0;
  asm volatile("tensor_load_to_lds %0, %1" :: "s"(g0), "s"(g1) : "memory");
}

// ------------------------------------------------------------------------

__global__ void cvt_f32_bf16(const float* __restrict__ src,
                             bf16* __restrict__ dst, int n) {
  int i = (blockIdx.x * blockDim.x + threadIdx.x) * 8;
  if (i + 7 < n) {
    float4 a = *(const float4*)(src + i);
    float4 b = *(const float4*)(src + i + 4);
    v8bf o;
    o[0] = (bf16)a.x; o[1] = (bf16)a.y; o[2] = (bf16)a.z; o[3] = (bf16)a.w;
    o[4] = (bf16)b.x; o[5] = (bf16)b.y; o[6] = (bf16)b.z; o[7] = (bf16)b.w;
    *(v8bf*)(dst + i) = o;
  }
}

// Y[m,n] = sum_k A[m,k] * W[n,k] + bias[n]
// MODE 0: write bf16 into [B,H,S,D];  MODE 1: write f32 into [B,S,E] (d_out)
template <int MODE>
__global__ __launch_bounds__(256)
void gemm_bf16(const bf16* __restrict__ A, const bf16* __restrict__ W,
               const float* __restrict__ bias, void* __restrict__ out) {
  const int lane = threadIdx.x & 31;
  const int wave = threadIdx.x >> 5;
  const int wm = wave & 3, wn = wave >> 2;           // 4 x 2 wave grid
  const int Mbase = blockIdx.y * 128 + wm * 32;
  const int Nbase = blockIdx.x * 128 + wn * 64;

  v8f acc[2][4];
#pragma unroll
  for (int mt = 0; mt < 2; ++mt)
#pragma unroll
    for (int nt = 0; nt < 4; ++nt)
#pragma unroll
      for (int v = 0; v < 8; ++v) acc[mt][nt][v] = 0.f;

  for (int k0 = 0; k0 < E; k0 += 32) {
    __builtin_prefetch(A + (Mbase + (lane & 15)) * E + k0 + 64, 0, 1);
    __builtin_prefetch(W + (Nbase + (lane & 15)) * E + k0 + 64, 0, 1);
    v16bf a0 = load_frag(A, E, Mbase, k0);
    v16bf a1 = load_frag(A, E, Mbase + 16, k0);
    v16bf bf[4];
#pragma unroll
    for (int nt = 0; nt < 4; ++nt) bf[nt] = load_frag(W, E, Nbase + nt * 16, k0);
#pragma unroll
    for (int nt = 0; nt < 4; ++nt) {
      acc[0][nt] = wmma_bf16(a0, bf[nt], acc[0][nt]);
      acc[1][nt] = wmma_bf16(a1, bf[nt], acc[1][nt]);
    }
  }

  const int half = lane >> 4, lanelo = lane & 15;
#pragma unroll
  for (int mt = 0; mt < 2; ++mt)
#pragma unroll
    for (int nt = 0; nt < 4; ++nt) {
      const int n = Nbase + nt * 16 + lanelo;
      const float bv = bias[n];
#pragma unroll
      for (int v = 0; v < 8; ++v) {
        const int m = Mbase + mt * 16 + v + half * 8;
        float val = acc[mt][nt][v] + bv;
        if (MODE == 0) {
          const int b = m >> 11, s = m & 2047, h = n >> 6, d = n & 63;
          ((bf16*)out)[((b * NH + h) * SEQ + s) * HD + d] = (bf16)val;
        } else {
          ((float*)out)[m * E + n] = val;
        }
      }
    }
}

// Flash attention: block = 4 waves, 64 query rows per block, one (b,h).
// K tile staged via TDM, V tile via async global->LDS, then LDS transpose.
__global__ __launch_bounds__(128)
void attention(const bf16* __restrict__ Q, const bf16* __restrict__ K,
               const bf16* __restrict__ V, bf16* __restrict__ O) {
  __shared__ bf16 ldsK[64 * 64];      // [kv][d]    (TDM dest)
  __shared__ bf16 ldsV[64 * 64];      // [kv][d]    (async dest)
  __shared__ bf16 ldsVt[64 * 64];     // [d][kv]    (transposed in LDS)
  __shared__ bf16 ldsP[4 * 16 * 64];  // per-wave P staging [16 q][64 kv]

  const int lane = threadIdx.x & 31;
  const int wave = threadIdx.x >> 5;
  const int half = lane >> 4, lanelo = lane & 15;
  const int qb = blockIdx.x, h = blockIdx.y, b = blockIdx.z;
  const int slab = ((b * NH + h) * SEQ) * HD;
  const bf16* Qb = Q + slab;
  const bf16* Kb = K + slab;
  const bf16* Vb = V + slab;

  const unsigned int ldsK_off = (unsigned int)(unsigned long long)(void*)&ldsK[0];
  const unsigned int ldsV_off = (unsigned int)(unsigned long long)(void*)&ldsV[0];

  // Q fragments (16 rows of this wave), with 1/sqrt(D)=0.125 folded in.
  const int qrow0 = qb * 64 + wave * 16;
  v16bf qf[2];
#pragma unroll
  for (int kk = 0; kk < 2; ++kk) {
    v16bf t = load_frag(Qb, HD, qrow0, kk * 32);
#pragma unroll
    for (int i = 0; i < 16; ++i) t[i] = (bf16)((float)t[i] * 0.125f);
    qf[kk] = t;
  }

  v8f o[4];
#pragma unroll
  for (int nt = 0; nt < 4; ++nt)
#pragma unroll
    for (int v = 0; v < 8; ++v) o[nt][v] = 0.f;
  float mrow[8], lrow[8];
#pragma unroll
  for (int v = 0; v < 8; ++v) { mrow[v] = -1e30f; lrow[v] = 0.f; }

  for (int kv0 = 0; kv0 < SEQ; kv0 += 64) {
    // --- stage K tile via Tensor Data Mover (wave 0 issues the DMA) ---
    if (wave == 0)
      tdm_load_2d(ldsK_off, Kb + kv0 * HD, HD, (unsigned)SEQ, HD, 64, HD);
    // --- stage V tile (row-major) via async global->LDS b128 copies ---
#pragma unroll
    for (int it = 0; it < 4; ++it) {
      int chunk = threadIdx.x + it * 128;          // 0..511
      int row = chunk >> 3, col8 = (chunk & 7) * 8;
      async_load_b128(ldsV_off + (unsigned)(row * 64 + col8) * 2,
                      (unsigned)(((kv0 + row) * HD + col8) * 2),
                      (unsigned long long)(const void*)Vb);
    }
    wait_asynccnt0();
    if (wave == 0) __builtin_amdgcn_s_wait_tensorcnt(0);
    __syncthreads();

    // --- transpose V within LDS: ldsVt[d][kv] = ldsV[kv][d] ---
#pragma unroll
    for (int it = 0; it < 4; ++it) {
      int chunk = threadIdx.x + it * 128;
      int row = chunk >> 3, col8 = (chunk & 7) * 8;
      v8bf vv = *(const v8bf*)(&ldsV[row * 64 + col8]);
#pragma unroll
      for (int j = 0; j < 8; ++j) ldsVt[(col8 + j) * 64 + row] = vv[j];
    }
    __syncthreads();

    // scores S = (Q*0.125) K^T : B-frag = rows of K tile
    v8f s[4];
#pragma unroll
    for (int nt = 0; nt < 4; ++nt) {
      v8f c;
#pragma unroll
      for (int v = 0; v < 8; ++v) c[v] = 0.f;
      c = wmma_bf16(qf[0], load_frag(ldsK, 64, nt * 16, 0), c);
      c = wmma_bf16(qf[1], load_frag(ldsK, 64, nt * 16, 32), c);
      s[nt] = c;
    }

    // online softmax row stats (rows split per half-wave; xor<16 stays in half)
    float newm[8], alpha[8], tsum[8];
#pragma unroll
    for (int v = 0; v < 8; ++v) {
      float t = fmaxf(fmaxf(s[0][v], s[1][v]), fmaxf(s[2][v], s[3][v]));
#pragma unroll
      for (int off = 1; off < 16; off <<= 1)
        t = fmaxf(t, __shfl_xor(t, off, 32));
      newm[v] = fmaxf(mrow[v], t);
      alpha[v] = __expf(mrow[v] - newm[v]);
      tsum[v] = 0.f;
    }

    bf16* pst = &ldsP[wave * 16 * 64];
#pragma unroll
    for (int nt = 0; nt < 4; ++nt)
#pragma unroll
      for (int v = 0; v < 8; ++v) {
        float p = __expf(s[nt][v] - newm[v]);
        tsum[v] += p;
        pst[(v + half * 8) * 64 + nt * 16 + lanelo] = (bf16)p;
      }
#pragma unroll
    for (int v = 0; v < 8; ++v) {
#pragma unroll
      for (int off = 1; off < 16; off <<= 1) tsum[v] += __shfl_xor(tsum[v], off, 32);
      lrow[v] = lrow[v] * alpha[v] + tsum[v];
      mrow[v] = newm[v];
    }

    // O = O*alpha + P V : A-frag = P from LDS, B-frag = rows of V^T
    v16bf pf0 = load_frag(pst, 64, 0, 0);
    v16bf pf1 = load_frag(pst, 64, 0, 32);
#pragma unroll
    for (int nt = 0; nt < 4; ++nt) {
#pragma unroll
      for (int v = 0; v < 8; ++v) o[nt][v] *= alpha[v];
      o[nt] = wmma_bf16(pf0, load_frag(ldsVt, 64, nt * 16, 0), o[nt]);
      o[nt] = wmma_bf16(pf1, load_frag(ldsVt, 64, nt * 16, 32), o[nt]);
    }
    __syncthreads();
  }

  // write normalized O to [B,S,E] bf16
#pragma unroll
  for (int nt = 0; nt < 4; ++nt)
#pragma unroll
    for (int v = 0; v < 8; ++v) {
      const int m = b * SEQ + qrow0 + v + half * 8;
      const int col = h * HD + nt * 16 + lanelo;
      O[m * E + col] = (bf16)(o[nt][v] / lrow[v]);
    }
}

extern "C" void kernel_launch(void* const* d_in, const int* in_sizes, int n_in,
                              void* d_out, int out_size, void* d_ws, size_t ws_size,
                              hipStream_t stream) {
  const float* x  = (const float*)d_in[0];
  const float* Wq = (const float*)d_in[1];
  const float* bq = (const float*)d_in[2];
  const float* Wk = (const float*)d_in[3];
  const float* bk = (const float*)d_in[4];
  const float* Wv = (const float*)d_in[5];
  const float* bv = (const float*)d_in[6];
  const float* Wo = (const float*)d_in[7];
  const float* bo = (const float*)d_in[8];

  char* ws = (char*)d_ws;
  bf16* Xb  = (bf16*)(ws);
  bf16* Wqb = (bf16*)(ws + (16u << 20));
  bf16* Wkb = (bf16*)(ws + (18u << 20));
  bf16* Wvb = (bf16*)(ws + (20u << 20));
  bf16* Wob = (bf16*)(ws + (22u << 20));
  bf16* Qt  = (bf16*)(ws + (24u << 20));
  bf16* Kt  = (bf16*)(ws + (40u << 20));
  bf16* Vt  = (bf16*)(ws + (56u << 20));
  bf16* Ob  = (bf16*)(ws + (72u << 20));

  const int nX = MTOT * E;   // 8388608
  const int nW = E * E;      // 1048576
  cvt_f32_bf16<<<nX / (8 * 256), 256, 0, stream>>>(x, Xb, nX);
  cvt_f32_bf16<<<nW / (8 * 256), 256, 0, stream>>>(Wq, Wqb, nW);
  cvt_f32_bf16<<<nW / (8 * 256), 256, 0, stream>>>(Wk, Wkb, nW);
  cvt_f32_bf16<<<nW / (8 * 256), 256, 0, stream>>>(Wv, Wvb, nW);
  cvt_f32_bf16<<<nW / (8 * 256), 256, 0, stream>>>(Wo, Wob, nW);

  dim3 ggrid(E / 128, MTOT / 128);  // (8, 64)
  gemm_bf16<0><<<ggrid, 256, 0, stream>>>(Xb, Wqb, bq, Qt);
  gemm_bf16<0><<<ggrid, 256, 0, stream>>>(Xb, Wkb, bk, Kt);
  gemm_bf16<0><<<ggrid, 256, 0, stream>>>(Xb, Wvb, bv, Vt);

  dim3 agrid(SEQ / 64, NH, BATCH);
  attention<<<agrid, 128, 0, stream>>>(Qt, Kt, Vt, Ob);

  gemm_bf16<1><<<ggrid, 256, 0, stream>>>(Ob, Wob, bo, (float*)d_out);
}
